// TwoSimplicialAttentionReference_2860448219223
// MI455X (gfx1250) — matrix operand order repaired and checked by reading, not compiled, not run on hardware
//
#include <hip/hip_runtime.h>
#include <math.h>

// Two-simplicial windowed attention, MI455X (gfx1250, wave32, WMMA + async-LDS).
// B=2, S=2048, H=4, D=64, W1=W2=64 -> window length A=C=65, padded to 80.

#define B_    2
#define S_    2048
#define H_    4
#define D_    64
#define W_    64
#define A_    65      // window length (w+1)
#define AP    80      // padded to 5 tiles of 16
#define KP    96      // padded K for GEMM2 (w columns)
#define SCALE 0.125f  // 1/sqrt(64)
#define L2E   1.4426950408889634f

typedef __attribute__((ext_vector_type(16))) _Float16 v16h;
typedef __attribute__((ext_vector_type(8)))  _Float16 v8h;
typedef __attribute__((ext_vector_type(8)))  float    v8f;

// ---- LDS layout (bytes), one 68624-byte arena ----
// pA   [AP][D_]  f16 :     0 .. 10240   A of GEMM1: q*k1*SCALE*log2e
// k2B  [AP][D_]  f16 : 10240 .. 20480   B of GEMM1: k2 window, [c][d]
// wW   [AP][KP]  f16 : 20480 .. 35840   A of GEMM2: exp2 weights, [a][c]
// v2t  [D_][KP]  f16 : 35840 .. 48128   B of GEMM2: v2 transposed, [d][c]
// v1s  [AP][D_]  f32 : 48128 .. 68608   final reduction operand (async-copied)
// dden [4]       f32 : 68608 .. 68624   per-wave denominator partials
#define SMEM_BYTES 68624
#define ZERO_OFF   20480            // zero wW, v2t, v1s, dden
#define ZERO_CH16  3009             // (68624-20480)/16

__device__ __forceinline__ v16h cat8(v8h lo, v8h hi) {
  return __builtin_shufflevector(lo, hi, 0,1,2,3,4,5,6,7,8,9,10,11,12,13,14,15);
}

// A fragment (16-bit A 16x32, ISA 7.12.2): lanes 0-15 / 16-31 hold the SAME
// row M=lane&15; hh=lane>>4 selects the K-half:
// halves[0..7]=K[kb+8hh .. +7], halves[8..15]=K[kb+16+8hh .. +7].
__device__ __forceinline__ v16h load_fragA(const _Float16* row, int kb, int hh) {
  v8h lo = *(const v8h*)(row + kb + 8 * hh);
  v8h hi = *(const v8h*)(row + kb + 16 + 8 * hh);
  return cat8(lo, hi);
}

// B fragment (16-bit B 32x16): lane holds column N=lane&15; same K order as A:
// halves[0..7]=K[kb+8hh .. +7], halves[8..15]=K[kb+16+8hh .. +7].
__device__ __forceinline__ v16h load_fragB(const _Float16* row, int kb, int hh) {
  v8h lo = *(const v8h*)(row + kb + 8 * hh);
  v8h hi = *(const v8h*)(row + kb + 16 + 8 * hh);
  return cat8(lo, hi);
}

__global__ __launch_bounds__(128)
void twosimp_attn_kernel(const float* __restrict__ q,
                         const float* __restrict__ k1,
                         const float* __restrict__ k2,
                         const float* __restrict__ v1,
                         const float* __restrict__ v2,
                         float* __restrict__ out)
{
  __shared__ __align__(16) unsigned char smem[SMEM_BYTES];
  _Float16* pA   = (_Float16*)(smem);
  _Float16* k2B  = (_Float16*)(smem + 10240);
  _Float16* wW   = (_Float16*)(smem + 20480);
  _Float16* v2t  = (_Float16*)(smem + 35840);
  float*    v1s  = (float*)   (smem + 48128);
  float*    dden = (float*)   (smem + 68608);

  const int tid  = threadIdx.x;
  const int lane = tid & 31;
  const int wv   = tid >> 5;            // wave id 0..3
  const int bsh  = blockIdx.x;          // ((b*S + s)*H + h)
  const int h    = bsh % H_;
  const int s    = (bsh / H_) % S_;
  const int b    = bsh / (H_ * S_);

  const long rowStride = (long)H_ * D_;
  const long base_bh   = ((long)b * S_ * H_ + h) * D_;   // element [b,0,h,0]

  // ---------------- Phase A0: zero pad regions ----------------------------
  {
    float4 z = make_float4(0.f, 0.f, 0.f, 0.f);
    for (int i = tid; i < ZERO_CH16; i += 128)
      *(float4*)(smem + ZERO_OFF + i * 16) = z;
  }
  __syncthreads();   // zero-stores (cross-wave) must land before staging writes

  // ---------------- Phase A1: stage windows into LDS ----------------------
  const int d0 = lane * 2;                      // each lane covers 2 d's
  float2 qv = *(const float2*)(q + base_bh + (long)s * rowStride + d0);
  const float qs = SCALE * L2E;                 // fold log2(e): WMMA yields score*log2e
  qv.x *= qs; qv.y *= qs;

  for (int a = wv; a < A_; a += 4) {            // window rows split across waves
    int j = s - W_ + a;                         // wave-uniform predicate
    if (j >= 0) {
      long ro = base_bh + (long)j * rowStride + d0;
      // v1 is a pure f32 pass-through: use the CDNA5 async global->LDS engine.
      // Low 32 bits of a generic LDS pointer == offset within the allocation.
      unsigned lds_off = (unsigned)(uintptr_t)(v1s + a * D_ + d0);
      asm volatile("global_load_async_to_lds_b64 %0, %1, off"
                   :: "v"(lds_off), "v"(v1 + ro) : "memory");

      float2 x1 = *(const float2*)(k1 + ro);
      float2 x2 = *(const float2*)(k2 + ro);
      float2 y2 = *(const float2*)(v2 + ro);
      pA [a * D_ + d0]     = (_Float16)(qv.x * x1.x);
      pA [a * D_ + d0 + 1] = (_Float16)(qv.y * x1.y);
      k2B[a * D_ + d0]     = (_Float16)x2.x;
      k2B[a * D_ + d0 + 1] = (_Float16)x2.y;
      v2t[(d0    ) * KP + a] = (_Float16)y2.x;  // transposed for GEMM2 B
      v2t[(d0 + 1) * KP + a] = (_Float16)y2.y;
    } else {
      // invalid (pre-window) rows: zero A/B operands; v2t/v1s already zero
      pA [a * D_ + d0]     = (_Float16)0.f;
      pA [a * D_ + d0 + 1] = (_Float16)0.f;
      k2B[a * D_ + d0]     = (_Float16)0.f;
      k2B[a * D_ + d0 + 1] = (_Float16)0.f;
    }
  }
  asm volatile("s_wait_asynccnt 0x0" ::: "memory");  // v1s resident before barrier
  __syncthreads();

  // ---------------- Phase B: scores*log2e = pA * k2B^T ; w = masked exp2 --
  const int hh   = lane >> 4;
  const int nn   = lane & 15;
  const int aLow = (s >= W_) ? 0 : (W_ - s);    // first valid window index
  float dsum = 0.f;

  for (int t = wv; t < 25; t += 4) {            // 5x5 score tiles
    const int a0 = (t / 5) * 16;
    const int c0 = (t % 5) * 16;
    const _Float16* arow = pA  + (a0 + nn) * D_;
    const _Float16* brow = k2B + (c0 + nn) * D_;

    v8f acc = {};
    acc = __builtin_amdgcn_wmma_f32_16x16x32_f16(false, load_fragA(arow,  0, hh),
                                                 false, load_fragB(brow,  0, hh),
                                                 (short)0, acc, false, false);
    acc = __builtin_amdgcn_wmma_f32_16x16x32_f16(false, load_fragA(arow, 32, hh),
                                                 false, load_fragB(brow, 32, hh),
                                                 (short)0, acc, false, false);

    const int  c   = c0 + nn;
    const bool cOK = (c >= aLow) && (c <= W_);
#pragma unroll
    for (int r = 0; r < 8; ++r) {               // C layout: a = a0 + r + 8*hh
      int   a  = a0 + r + 8 * hh;
      bool  ok = cOK && (a >= aLow) && (a <= W_);
      // raw v_exp_f32 (native exp2): branch-free, log2e pre-folded into scores
      float e   = __builtin_amdgcn_exp2f(acc[r]);
      float wv_ = ok ? e : 0.f;
      dsum += wv_;
      wW[a * KP + c] = (_Float16)wv_;           // cols 80..95 pre-zeroed
    }
  }
#pragma unroll
  for (int off = 16; off; off >>= 1) dsum += __shfl_xor(dsum, off, 32);
  if (lane == 0) dden[wv] = dsum;
  __syncthreads();

  // ---------------- Phase C: tmp = wW * v2t^T ; num[d] = sum_a tmp*v1 -----
  const int dcol = wv * 16;                     // each wave owns one d-tile
  float num = 0.f;
#pragma unroll
  for (int ti = 0; ti < 5; ++ti) {
    const int a0 = ti * 16;
    const _Float16* arow = wW  + (a0   + nn) * KP;
    const _Float16* brow = v2t + (dcol + nn) * KP;
    v8f acc = {};
#pragma unroll
    for (int kb = 0; kb < KP; kb += 32) {       // K = 96 (zero padded)
      acc = __builtin_amdgcn_wmma_f32_16x16x32_f16(false, load_fragA(arow, kb, hh),
                                                   false, load_fragB(brow, kb, hh),
                                                   (short)0, acc, false, false);
    }
#pragma unroll
    for (int r = 0; r < 8; ++r) {               // multiply by v1 in-register
      int a = a0 + r + 8 * hh;                  // padded rows: tmp==0 and v1s==0
      num += acc[r] * v1s[a * D_ + dcol + nn];
    }
  }
  num += __shfl_xor(num, 16, 32);               // combine the two a-halves

  const float denom = dden[0] + dden[1] + dden[2] + dden[3] + 1e-8f;
  if (lane < 16) {
    out[(long)bsh * D_ + dcol + lane] = num / denom;
  }
}

extern "C" void kernel_launch(void* const* d_in, const int* in_sizes, int n_in,
                              void* d_out, int out_size, void* d_ws, size_t ws_size,
                              hipStream_t stream) {
  (void)in_sizes; (void)n_in; (void)d_ws; (void)ws_size; (void)out_size;
  const float* q  = (const float*)d_in[0];
  const float* k1 = (const float*)d_in[1];
  const float* k2 = (const float*)d_in[2];
  const float* v1 = (const float*)d_in[3];
  const float* v2 = (const float*)d_in[4];
  float* out = (float*)d_out;

  dim3 grid(B_ * S_ * H_);   // 16384 blocks, one per (b,s,h)
  dim3 block(128);           // 4 waves: tile-parallel GEMMs + in-register reductions
  twosimp_attn_kernel<<<grid, block, 0, stream>>>(q, k1, k2, v1, v2, out);
}
